// GenericRNNSequenceEncoder_15298673508809
// MI455X (gfx1250) — matrix operand order repaired and checked by reading, not compiled, not run on hardware
//
#include <hip/hip_runtime.h>
#include <hip/hip_bf16.h>

// Problem dims (fixed by the reference).
#define B_  32
#define T_  512
#define D_  512
#define H_  1024
#define G_  4096     // 4*H (gate width)
#define KT_ 1536     // D + H (fused K)
#define LDA_ (KT_ + 8)  // padded LDS row stride (bf16 elems): 772 dwords % 64 = 4
                        // -> consecutive rows hit different banks, 16B aligned.

typedef __attribute__((ext_vector_type(16))) __bf16        v16bf;
typedef __attribute__((ext_vector_type(8)))  float         v8f;
typedef __attribute__((ext_vector_type(4)))  unsigned int  v4u;

// Matches the async-to-LDS builtin's parameter type exactly:
// pointer to 16B int vector in global (AS1) / local (AS3) address space.
typedef int v4i_ __attribute__((vector_size(16)));
typedef __attribute__((address_space(1))) v4i_* g_v4i;
typedef __attribute__((address_space(3))) v4i_* l_v4i;

union Frag32B { v4u u[2]; v16bf v; };

__device__ __forceinline__ unsigned short f2bf(float f) {
  union { float f; unsigned u; } x; x.f = f;
  unsigned r = x.u + 0x7FFFu + ((x.u >> 16) & 1u);   // round-to-nearest-even
  return (unsigned short)(r >> 16);
}
__device__ __forceinline__ float sigm(float x) { return 1.0f / (1.0f + __expf(-x)); }

// ---- one-time prep kernels --------------------------------------------------

__global__ void k_convert_inputs(const float* __restrict__ x,
                                 unsigned short* __restrict__ xb, int n) {
  int i = blockIdx.x * blockDim.x + threadIdx.x;
  if (i < n) xb[i] = f2bf(x[i]);
}

// wcatT[n][k] = bf16( k<D ? Wi[k][n] : Wh[k-D][n] )  — B transposed so each
// output column n has contiguous K (matches the WMMA B-fragment layout).
__global__ void k_build_wcat(const float* __restrict__ Wi,
                             const float* __restrict__ Wh,
                             unsigned short* __restrict__ wc, int n) {
  int i = blockIdx.x * blockDim.x + threadIdx.x;
  if (i >= n) return;
  int nn = i / KT_, k = i % KT_;
  float v = (k < D_) ? Wi[(size_t)k * G_ + nn] : Wh[(size_t)(k - D_) * G_ + nn];
  wc[i] = f2bf(v);
}

__global__ void k_init_state(const float* __restrict__ c0,
                             const float* __restrict__ h0,
                             float* __restrict__ c,
                             unsigned short* __restrict__ hb, int n) {
  int i = blockIdx.x * blockDim.x + threadIdx.x;
  if (i < n) { c[i] = c0[i]; hb[i] = f2bf(h0[i]); }
}

// ---- per-step gates GEMM: gates[32][4096] = [x_t || h] @ wcatT^T ------------
// 64 blocks x 4 waves. A (32x1536 bf16, 96KB) staged ONCE per block into LDS
// via async global->LDS; B (L2-resident, 12.6MB bf16) streamed with manual
// double-buffering + prefetch. Each wave owns one 16-col N tile, both M tiles.

__global__ __launch_bounds__(128)
void k_gates_gemm(const unsigned short* __restrict__ xb,   // [B][T][D] bf16
                  const unsigned short* __restrict__ hb,   // [B][H]    bf16
                  const unsigned short* __restrict__ wc,   // [4H][D+H] bf16
                  float* __restrict__ gates,               // [B][4H]   f32
                  int t)
{
  extern __shared__ unsigned short sA[];   // [32][LDA_] bf16, ~96.5 KB

  const int wave  = threadIdx.x >> 5;
  const int lane  = threadIdx.x & 31;
  const int lhalf = lane >> 4;          // 0/1: K-half per ISA fragment layout
  const int lrow  = lane & 15;          // row (A) / col (B) within 16-tile
  const int ntile = (blockIdx.x << 2) + wave;   // 0..255
  const int ncol  = ntile << 4;

  // ---- stage A = [x_t || h] into LDS (async copy, 16B chunks) ----
  for (int c = threadIdx.x; c < B_ * (KT_ / 8); c += 128) {
    const int row = c / (KT_ / 8);
    const int k   = (c % (KT_ / 8)) * 8;
    const unsigned short* src = (k < D_)
        ? xb + (((size_t)row * T_) + t) * D_ + k
        : hb + (size_t)row * H_ + (k - D_);
    unsigned short* dst = sA + row * LDA_ + k;
#if __has_builtin(__builtin_amdgcn_global_load_async_to_lds_b128)
    __builtin_amdgcn_global_load_async_to_lds_b128(
        (g_v4i)(void*)src, (l_v4i)dst, 0, 0);
#else
    *(v4u*)dst = *(const v4u*)src;     // global_load_b128 + ds_store_b128
#endif
  }
#if __has_builtin(__builtin_amdgcn_global_load_async_to_lds_b128)
#if __has_builtin(__builtin_amdgcn_s_wait_asynccnt)
  __builtin_amdgcn_s_wait_asynccnt(0);
#else
  asm volatile("s_wait_asynccnt 0" ::: "memory");
#endif
#endif
  __syncthreads();

  // ---- K loop: A from LDS, B from global (double-buffered) ----
  const unsigned short* bp  = wc + (size_t)(ncol + lrow) * KT_ + (lhalf << 3);
  const int kaoff = lhalf << 3;
  const unsigned short* la0 = sA + (size_t)lrow        * LDA_ + kaoff;
  const unsigned short* la1 = sA + (size_t)(lrow + 16) * LDA_ + kaoff;

  v8f acc0 = {};
  v8f acc1 = {};

  Frag32B bcur;
  bcur.u[0] = *(const v4u*)(bp);
  bcur.u[1] = *(const v4u*)(bp + 16);

  for (int k0 = 0; k0 < KT_; k0 += 32) {
    // prefetch next B fragment while WMMAs consume the current one
    Frag32B bnxt;
    const unsigned short* pn = bp + ((k0 + 32 < KT_) ? (k0 + 32) : 0);
    bnxt.u[0] = *(const v4u*)(pn);
    bnxt.u[1] = *(const v4u*)(pn + 16);
    __builtin_prefetch((const void*)(pn + 32), 0, 0);   // global_prefetch_b8

    Frag32B a0, a1;
    a0.u[0] = *(const v4u*)(la0 + k0);  a0.u[1] = *(const v4u*)(la0 + k0 + 16);
    a1.u[0] = *(const v4u*)(la1 + k0);  a1.u[1] = *(const v4u*)(la1 + k0 + 16);

    acc0 = __builtin_amdgcn_wmma_f32_16x16x32_bf16(false, a0.v, false, bcur.v,
                                                   (short)0, acc0, false, false);
    acc1 = __builtin_amdgcn_wmma_f32_16x16x32_bf16(false, a1.v, false, bcur.v,
                                                   (short)0, acc1, false, false);
    bcur = bnxt;
  }

  // C/D layout: VGPR v holds M = v + 8*lane_half, N = lane&15.
  #pragma unroll
  for (int v = 0; v < 8; ++v) {
    const int m = v + (lhalf << 3);
    gates[(size_t)m        * G_ + ncol + lrow] = acc0[v];
    gates[(size_t)(m + 16) * G_ + ncol + lrow] = acc1[v];
  }
}

// ---- per-step LSTM cell (fp32 state), writes outputs + final-state gather ---

__global__ __launch_bounds__(256)
void k_cell(const float* __restrict__ gates, const float* __restrict__ bias,
            const int* __restrict__ lengths, float* __restrict__ c,
            unsigned short* __restrict__ hb, float* __restrict__ out, int t)
{
  int i = blockIdx.x * blockDim.x + threadIdx.x;
  if (i >= B_ * H_) return;
  int b = i >> 10;            // H_ = 1024
  int j = i & (H_ - 1);
  const float* gr = gates + (size_t)b * G_;
  float gi = gr[j]            + bias[j];
  float gf = gr[H_ + j]       + bias[H_ + j];
  float gg = gr[2 * H_ + j]   + bias[2 * H_ + j];
  float go = gr[3 * H_ + j]   + bias[3 * H_ + j];
  float cn = sigm(gf) * c[i] + sigm(gi) * tanhf(gg);
  float hn = sigm(go) * tanhf(cn);
  c[i]  = cn;
  hb[i] = f2bf(hn);                                  // bf16 h for next step GEMM
  out[((size_t)b * T_ + t) * H_ + j] = hn;           // outputs[b][t][j]
  if (t == lengths[b] - 1) {                         // final-state gather
    out[(size_t)B_ * T_ * H_ + i]           = cn;    // c_final
    out[(size_t)B_ * T_ * H_ + B_ * H_ + i] = hn;    // h_final
  }
}

// ---- launch -----------------------------------------------------------------

extern "C" void kernel_launch(void* const* d_in, const int* in_sizes, int n_in,
                              void* d_out, int out_size, void* d_ws, size_t ws_size,
                              hipStream_t stream) {
  (void)in_sizes; (void)n_in; (void)out_size; (void)ws_size;
  const float* inputs  = (const float*)d_in[0];
  const int*   lengths = (const int*)  d_in[1];
  const float* c0      = (const float*)d_in[2];
  const float* h0      = (const float*)d_in[3];
  const float* Wi      = (const float*)d_in[4];
  const float* Wh      = (const float*)d_in[5];
  const float* bias    = (const float*)d_in[6];
  float* out = (float*)d_out;

  // Workspace layout (all 16B aligned):
  //  xb   : B*T*D bf16      16,777,216 B
  //  wc   : 4H*(D+H) bf16   12,582,912 B
  //  hb   : B*H bf16            65,536 B
  //  c_ws : B*H f32            131,072 B
  //  gates: B*4H f32           524,288 B   (total ~30.1 MB)
  char* ws = (char*)d_ws;
  unsigned short* xb    = (unsigned short*)(ws);
  unsigned short* wc    = (unsigned short*)(ws + 16777216);
  unsigned short* hb    = (unsigned short*)(ws + 16777216 + 12582912);
  float*          c_ws  = (float*)(ws + 16777216 + 12582912 + 65536);
  float*          gates = (float*)(ws + 16777216 + 12582912 + 65536 + 131072);

  const int nx = B_ * T_ * D_;
  k_convert_inputs<<<(nx + 255) / 256, 256, 0, stream>>>(inputs, xb, nx);
  const int nw = G_ * KT_;
  k_build_wcat<<<(nw + 255) / 256, 256, 0, stream>>>(Wi, Wh, wc, nw);
  k_init_state<<<(B_ * H_ + 255) / 256, 256, 0, stream>>>(c0, h0, c_ws, hb, B_ * H_);

  const size_t lds_bytes = (size_t)B_ * LDA_ * sizeof(unsigned short); // ~96.5 KB
  for (int t = 0; t < T_; ++t) {
    k_gates_gemm<<<64, 128, lds_bytes, stream>>>(xb, hb, wc, gates, t);
    k_cell<<<(B_ * H_ + 255) / 256, 256, 0, stream>>>(gates, bias, lengths,
                                                      c_ws, hb, out, t);
  }
}
